// MultiHeadAttention_10677288698116
// MI455X (gfx1250) — matrix operand order repaired and checked by reading, not compiled, not run on hardware
//
#include <hip/hip_runtime.h>

typedef _Float16 v16h __attribute__((ext_vector_type(16)));
typedef _Float16 v8h  __attribute__((ext_vector_type(8)));
typedef _Float16 v4h  __attribute__((ext_vector_type(4)));
typedef float    v8f  __attribute__((ext_vector_type(8)));

#define CAT16(lo, hi) __builtin_shufflevector(lo, hi, 0,1,2,3,4,5,6,7,8,9,10,11,12,13,14,15)

static __device__ __forceinline__ v8f wmma16(v16h a, v16h b, v8f c) {
  // D = A(16x32 f16) * B(32x16 f16) + C(16x16 f32)
  return __builtin_amdgcn_wmma_f32_16x16x32_f16(false, a, false, b, (short)0, c, false, false);
}

static __device__ __forceinline__ v8f zero8() {
  v8f z = {0.f, 0.f, 0.f, 0.f, 0.f, 0.f, 0.f, 0.f};
  return z;
}

// ---------------------------------------------------------------------------
// Intra-16-lane XOR permute (all-reduce building block) via v_permlane16_b32:
// pure VALU, no LDS traffic / DScnt waits (unlike __shfl_xor -> ds_bpermute).
// ---------------------------------------------------------------------------
static constexpr unsigned mksel(int m, int base) {
  unsigned r = 0;
  for (int i = 0; i < 8; ++i) r |= (unsigned)(((base + i) ^ m) & 15) << (4 * i);
  return r;
}

template <int M>
static __device__ __forceinline__ float shf16(float v) {
#if __has_builtin(__builtin_amdgcn_permlane16)
  constexpr unsigned s0 = mksel(M, 0);
  constexpr unsigned s1 = mksel(M, 8);
  unsigned u = __float_as_uint(v);
  return __uint_as_float(__builtin_amdgcn_permlane16(u, u, s0, s1, false, false));
#else
  return __shfl_xor(v, M, 32);
#endif
}

static __device__ __forceinline__ float redmax16(float v) {
  v = fmaxf(v, shf16<1>(v));
  v = fmaxf(v, shf16<2>(v));
  v = fmaxf(v, shf16<4>(v));
  v = fmaxf(v, shf16<8>(v));
  return v;
}
static __device__ __forceinline__ float redsum16(float v) {
  v += shf16<1>(v);
  v += shf16<2>(v);
  v += shf16<4>(v);
  v += shf16<8>(v);
  return v;
}

// ---------------------------------------------------------------------------
// CDNA5 async global->LDS copy (ASYNCcnt path).
// Flat LDS addresses: low 32 bits of a generic pointer into __shared__ ARE the
// LDS byte offset.
// ---------------------------------------------------------------------------
static __device__ __forceinline__ void async_ld16(const _Float16* g, unsigned lds) {
  asm volatile("global_load_async_to_lds_b128 %0, %1, off" :: "v"(lds), "v"(g) : "memory");
}
static __device__ __forceinline__ void wait_async0() {
#if __has_builtin(__builtin_amdgcn_s_wait_asynccnt)
  __builtin_amdgcn_s_wait_asynccnt(0);
#else
  asm volatile("s_wait_asynccnt 0x0" ::: "memory");
#endif
}
static __device__ __forceinline__ unsigned lds_off(const void* p) {
  return (unsigned)(uintptr_t)p;
}

// ---------------------------------------------------------------------------
// Stage 0a: x (f32, 8192x1024 row-major) -> f16, same layout
// ---------------------------------------------------------------------------
__global__ __launch_bounds__(256) void cvt_x_kernel(const float* __restrict__ x,
                                                    _Float16* __restrict__ xh) {
  size_t i = ((size_t)blockIdx.x * 256 + threadIdx.x) * 4;
  float4 v = *(const float4*)(x + i);
  v4h o = {(_Float16)v.x, (_Float16)v.y, (_Float16)v.z, (_Float16)v.w};
  *(v4h*)(xh + i) = o;
}

// ---------------------------------------------------------------------------
// Stage 0b: w (f32, K x N row-major) -> wT (f16, N x K row-major)
// ---------------------------------------------------------------------------
__global__ __launch_bounds__(256) void transpose_cvt(const float* __restrict__ w,
                                                     _Float16* __restrict__ wT) {
  __shared__ float tile[32][33];
  const int tx = threadIdx.x, ty = threadIdx.y;        // blockDim = (32, 8)
  const int n0 = blockIdx.x * 32, k0 = blockIdx.y * 32;
  #pragma unroll
  for (int i = 0; i < 4; ++i)
    tile[ty + 8 * i][tx] = w[(size_t)(k0 + ty + 8 * i) * 1024 + n0 + tx];
  __syncthreads();
  #pragma unroll
  for (int i = 0; i < 4; ++i)
    wT[(size_t)(n0 + ty + 8 * i) * 1024 + k0 + tx] = (_Float16)tile[tx][ty + 8 * i];
}

// ---------------------------------------------------------------------------
// Tiled WMMA GEMM: (8192x1024) x (1024x1024) + bias
//   A  : f16, M x K row-major
//   Bt : f16, N x K row-major (pre-transposed weight)
// Block tile 128x128x32, 256 threads = 8 waves, wave tile 32x64.
// LDS staging is double-buffered via GLOBAL_LOAD_ASYNC_TO_LDS_B128: tile k+1
// streams in (ASYNCcnt) while tile k feeds the WMMAs; 1 barrier per K-step.
// MODE 0: out16[(b,h,t,d)]   f16   (Q,K for attention)
// MODE 1: out16[(b,h,d,t)]   f16   (V transposed for attention PV GEMM)
// MODE 2: out32[m*N + n]     f32   (final projection to d_out)
// ---------------------------------------------------------------------------
template <int MODE>
__global__ __launch_bounds__(256) void gemm_wmma_k(
    const _Float16* __restrict__ A, const _Float16* __restrict__ Bt,
    const float* __restrict__ bias, _Float16* __restrict__ out16,
    float* __restrict__ out32) {
  constexpr int K = 1024, N = 1024;
  constexpr int LP = 40;  // padded LDS pitch (halfs): 80B rows, 16B-aligned chunks
  __shared__ _Float16 lA[2][128 * LP];
  __shared__ _Float16 lB[2][128 * LP];

  const int tid = threadIdx.x;
  const int wave = tid >> 5;
  const int lane = tid & 31;
  const int lm = lane & 15;
  const int g = lane >> 4;
  const int waveM = (wave & 3) * 32;
  const int waveN = (wave >> 2) * 64;
  const int bm = blockIdx.x * 128;
  const int bn = blockIdx.y * 128;

  // each thread deposits 2 chunks of 8 halfs per tile (rows r and r+64)
  const int crow = tid >> 2;
  const int ccol = (tid & 3) * 8;
  const _Float16* gA0 = A + (size_t)(bm + crow) * K + ccol;
  const _Float16* gA1 = A + (size_t)(bm + crow + 64) * K + ccol;
  const _Float16* gB0 = Bt + (size_t)(bn + crow) * K + ccol;
  const _Float16* gB1 = Bt + (size_t)(bn + crow + 64) * K + ccol;

  auto issue = [&](int buf, int k0) {
    async_ld16(gA0 + k0, lds_off(&lA[buf][crow * LP + ccol]));
    async_ld16(gA1 + k0, lds_off(&lA[buf][(crow + 64) * LP + ccol]));
    async_ld16(gB0 + k0, lds_off(&lB[buf][crow * LP + ccol]));
    async_ld16(gB1 + k0, lds_off(&lB[buf][(crow + 64) * LP + ccol]));
  };

  v8f acc[2][4];
  #pragma unroll
  for (int i = 0; i < 2; ++i)
    #pragma unroll
    for (int j = 0; j < 4; ++j) acc[i][j] = zero8();

  issue(0, 0);
  int buf = 0;
  for (int k0 = 0; k0 < K; k0 += 32, buf ^= 1) {
    wait_async0();     // this wave's async deposits have landed
    __syncthreads();   // ... and every other wave's too (their reads of buf^1 done)
    if (k0 + 32 < K) issue(buf ^ 1, k0 + 32);

    v16h af[2];
    #pragma unroll
    for (int i = 0; i < 2; ++i) {
      const _Float16* rp = &lA[buf][(waveM + i * 16 + lm) * LP];
      v8h c0 = *(const v8h*)(rp + g * 8);
      v8h c1 = *(const v8h*)(rp + 16 + g * 8);
      af[i] = CAT16(c0, c1);
    }
    v16h bf[4];
    #pragma unroll
    for (int j = 0; j < 4; ++j) {
      const _Float16* rp = &lB[buf][(waveN + j * 16 + lm) * LP];
      v8h c0 = *(const v8h*)(rp + g * 8);
      v8h c1 = *(const v8h*)(rp + 16 + g * 8);
      bf[j] = CAT16(c0, c1);
    }
    #pragma unroll
    for (int i = 0; i < 2; ++i)
      #pragma unroll
      for (int j = 0; j < 4; ++j) acc[i][j] = wmma16(af[i], bf[j], acc[i][j]);
  }

  #pragma unroll
  for (int j = 0; j < 4; ++j) {
    const int nc = bn + waveN + j * 16 + lm;
    const float bv = bias[nc];
    #pragma unroll
    for (int i = 0; i < 2; ++i) {
      #pragma unroll
      for (int r = 0; r < 8; ++r) {
        const int mr = bm + waveM + i * 16 + r + 8 * g;
        const float val = acc[i][j][r] + bv;
        if (MODE == 2) {
          out32[(size_t)mr * N + nc] = val;
        } else {
          const int b = mr >> 11, t = mr & 2047;   // T = 2048
          const int h = nc >> 6, d = nc & 63;      // Dh = 64
          if (MODE == 0)
            out16[((size_t)(b * 16 + h) * 2048 + t) * 64 + d] = (_Float16)val;
          else
            out16[((size_t)(b * 16 + h) * 64 + d) * 2048 + t] = (_Float16)val;
        }
      }
    }
  }
}

// ---------------------------------------------------------------------------
// Causal flash attention.  Q,K: (B*H, T, 64) f16.  Vt: (B*H, 64, T) f16.
// Grid (T/64, B*H), 128 threads = 4 waves; each wave owns 16 query rows.
// Row softmax stats live per half-wave (rows r+8g); reductions via permlane16.
// Output written f16 to (B, T, C) layout (A-matrix of the final GEMM).
// ---------------------------------------------------------------------------
__global__ __launch_bounds__(128) void attn_k(const _Float16* __restrict__ Q,
                                              const _Float16* __restrict__ Km,
                                              const _Float16* __restrict__ Vt,
                                              _Float16* __restrict__ O) {
  __shared__ _Float16 ldsP[4][16 * 72];  // per-wave P staging, 144B pitch
  const int tid = threadIdx.x;
  const int wave = tid >> 5;
  const int lane = tid & 31;
  const int lm = lane & 15;
  const int g = lane >> 4;
  const int bh = blockIdx.y;
  const int qb = blockIdx.x;
  const size_t base = (size_t)bh * 2048 * 64;
  const int q0 = qb * 64 + wave * 16;

  // Q A-fragments for the two Dh k-steps (Dh = 64 = 2 x 32)
  v16h qf[2];
  #pragma unroll
  for (int kk = 0; kk < 2; ++kk) {
    const _Float16* rp = Q + base + (size_t)(q0 + lm) * 64 + kk * 32;
    v8h c0 = *(const v8h*)(rp + g * 8);
    v8h c1 = *(const v8h*)(rp + 16 + g * 8);
    qf[kk] = CAT16(c0, c1);
  }

  float m[8], l[8];
  v8f oacc[4];
  #pragma unroll
  for (int r = 0; r < 8; ++r) { m[r] = -1e30f; l[r] = 0.f; }
  #pragma unroll
  for (int jd = 0; jd < 4; ++jd) oacc[jd] = zero8();

  const int nblk = qb + 1;  // causal: only key blocks <= query block
  for (int blk = 0; blk < nblk; ++blk) {
    const int kb = blk * 64;

    // S = Q * K^T  (16 x 64 per wave)
    v8f s[4];
    #pragma unroll
    for (int j = 0; j < 4; ++j) s[j] = zero8();
    #pragma unroll
    for (int j = 0; j < 4; ++j) {
      const int key = kb + j * 16 + lm;
      #pragma unroll
      for (int kk = 0; kk < 2; ++kk) {
        const _Float16* rp = Km + base + (size_t)key * 64 + kk * 32 + g * 8;
        v8h c0 = *(const v8h*)(rp);
        v8h c1 = *(const v8h*)(rp + 16);
        s[j] = wmma16(qf[kk], CAT16(c0, c1), s[j]);
      }
    }

    // scale + causal mask + per-row max
    float pm[8];
    #pragma unroll
    for (int r = 0; r < 8; ++r) pm[r] = -1e30f;
    #pragma unroll
    for (int j = 0; j < 4; ++j) {
      const int key = kb + j * 16 + lm;
      #pragma unroll
      for (int r = 0; r < 8; ++r) {
        const int q = q0 + r + 8 * g;
        float v = s[j][r] * 0.125f;  // 1/sqrt(64)
        if (key > q) v = -1e30f;
        s[j][r] = v;
        pm[r] = fmaxf(pm[r], v);
      }
    }
    #pragma unroll
    for (int r = 0; r < 8; ++r) pm[r] = redmax16(pm[r]);

    float alpha[8], rs[8];
    #pragma unroll
    for (int r = 0; r < 8; ++r) {
      const float mn = fmaxf(m[r], pm[r]);
      alpha[r] = __expf(m[r] - mn);
      m[r] = mn;
      rs[r] = 0.f;
    }

    __syncthreads();  // previous iteration's P reads are done
    #pragma unroll
    for (int j = 0; j < 4; ++j)
      #pragma unroll
      for (int r = 0; r < 8; ++r) {
        const float p = __expf(s[j][r] - m[r]);
        rs[r] += p;
        ldsP[wave][(r + 8 * g) * 72 + j * 16 + lm] = (_Float16)p;
      }
    #pragma unroll
    for (int r = 0; r < 8; ++r) rs[r] = redsum16(rs[r]);
    #pragma unroll
    for (int r = 0; r < 8; ++r) l[r] = l[r] * alpha[r] + rs[r];
    #pragma unroll
    for (int jd = 0; jd < 4; ++jd)
      #pragma unroll
      for (int r = 0; r < 8; ++r) oacc[jd][r] *= alpha[r];
    __syncthreads();  // P staged

    // O += P(16x64) * V(64x64): keys are the K dim (2 steps of 32)
    #pragma unroll
    for (int kk = 0; kk < 2; ++kk) {
      const _Float16* rp = &ldsP[wave][lm * 72 + kk * 32];
      v8h c0 = *(const v8h*)(rp + g * 8);
      v8h c1 = *(const v8h*)(rp + 16 + g * 8);
      const v16h pf = CAT16(c0, c1);
      #pragma unroll
      for (int jd = 0; jd < 4; ++jd) {
        const _Float16* vp =
            Vt + (size_t)(bh * 64 + jd * 16 + lm) * 2048 + kb + kk * 32 + g * 8;
        v8h d0 = *(const v8h*)(vp);
        v8h d1 = *(const v8h*)(vp + 16);
        oacc[jd] = wmma16(pf, CAT16(d0, d1), oacc[jd]);
      }
    }
  }

  // finalize: O /= l, write to (B, T, C) f16
  const int b = bh >> 4, h = bh & 15;
  #pragma unroll
  for (int jd = 0; jd < 4; ++jd)
    #pragma unroll
    for (int r = 0; r < 8; ++r) {
      const int q = q0 + r + 8 * g;
      const int d = jd * 16 + lm;
      O[(size_t)(b * 2048 + q) * 1024 + h * 64 + d] =
          (_Float16)(oacc[jd][r] / l[r]);
    }
}

// ---------------------------------------------------------------------------
extern "C" void kernel_launch(void* const* d_in, const int* in_sizes, int n_in,
                              void* d_out, int out_size, void* d_ws, size_t ws_size,
                              hipStream_t stream) {
  const float* x  = (const float*)d_in[0];
  const float* wq = (const float*)d_in[1];
  const float* bq = (const float*)d_in[2];
  const float* wk = (const float*)d_in[3];
  const float* bk = (const float*)d_in[4];
  const float* wv = (const float*)d_in[5];
  const float* bv = (const float*)d_in[6];
  const float* wo = (const float*)d_in[7];
  const float* bo = (const float*)d_in[8];
  float* out = (float*)d_out;

  char* ws = (char*)d_ws;
  const size_t SZ_X = (size_t)8192 * 1024 * sizeof(_Float16);  // 16 MiB
  const size_t SZ_W = (size_t)1024 * 1024 * sizeof(_Float16);  //  2 MiB
  _Float16* xh  = (_Float16*)(ws);
  _Float16* wqT = (_Float16*)(ws + SZ_X);
  _Float16* wkT = (_Float16*)(ws + SZ_X + 1 * SZ_W);
  _Float16* wvT = (_Float16*)(ws + SZ_X + 2 * SZ_W);
  _Float16* woT = (_Float16*)(ws + SZ_X + 3 * SZ_W);
  _Float16* Qh  = (_Float16*)(ws + 1 * SZ_X + 4 * SZ_W);
  _Float16* Kh  = (_Float16*)(ws + 2 * SZ_X + 4 * SZ_W);
  _Float16* Vth = (_Float16*)(ws + 3 * SZ_X + 4 * SZ_W);
  _Float16* attn16 = xh;  // xh is dead after the V projection: reuse it

  cvt_x_kernel<<<8192, 256, 0, stream>>>(x, xh);
  dim3 tb(32, 8), tg(32, 32);
  transpose_cvt<<<tg, tb, 0, stream>>>(wq, wqT);
  transpose_cvt<<<tg, tb, 0, stream>>>(wk, wkT);
  transpose_cvt<<<tg, tb, 0, stream>>>(wv, wvT);
  transpose_cvt<<<tg, tb, 0, stream>>>(wo, woT);

  dim3 gg(8192 / 128, 1024 / 128);
  gemm_wmma_k<0><<<gg, 256, 0, stream>>>(xh, wqT, bq, Qh, nullptr);
  gemm_wmma_k<0><<<gg, 256, 0, stream>>>(xh, wkT, bk, Kh, nullptr);
  gemm_wmma_k<1><<<gg, 256, 0, stream>>>(xh, wvT, bv, Vth, nullptr);

  attn_k<<<dim3(32, 64), 128, 0, stream>>>(Qh, Kh, Vth, attn16);

  gemm_wmma_k<2><<<gg, 256, 0, stream>>>(attn16, woT, bo, nullptr, out);

  (void)in_sizes; (void)n_in; (void)out_size; (void)ws_size;
}
